// GraphormerExpert_20538533609927
// MI455X (gfx1250) — matrix operand order repaired and checked by reading, not compiled, not run on hardware
//
#include <hip/hip_runtime.h>
#include <hip/hip_bf16.h>

typedef __attribute__((ext_vector_type(16))) _Float16 v16h;
typedef __attribute__((ext_vector_type(8)))  float    v8f;

#define PITCH 136   // f16 elements per LDS row (128 + 8 pad)

// ---------------- order-preserving float<->uint for atomicMax softmax max ----
__device__ __forceinline__ unsigned enc_f(float f) {
    unsigned u = __float_as_uint(f);
    return (u & 0x80000000u) ? ~u : (u | 0x80000000u);
}
__device__ __forceinline__ float dec_f(unsigned u) {
    return (u & 0x80000000u) ? __uint_as_float(u & 0x7FFFFFFFu) : __uint_as_float(~u);
}
#define ENC_NEG_INF 0x007FFFFFu   // enc_f(-inf)

// ---------------- fused (optional LN) + up-to-4-weight WMMA GEMM -------------
// OUTi[n,128] = LN?(A)[n,128] @ Wi[128,128] + bi   for i < nw
// Block: 256 threads = 8 wave32; block tile = 64 rows x 128 cols.
// Staging: wave w LayerNorms rows 8w..8w+7 into LDS (f16) once; then for each
// weight: stage W^T to LDS (prefetching the next weight), run 4x(16x16) WMMA
// tiles per wave over K=128 in 4 steps of 32.
__global__ __launch_bounds__(256) void fused_ln_gemm4_kernel(
    const float* __restrict__ A, int doLN,
    const float* __restrict__ lns, const float* __restrict__ lnb,
    const float* __restrict__ W0, const float* __restrict__ b0, float* __restrict__ O0,
    const float* __restrict__ W1, const float* __restrict__ b1, float* __restrict__ O1,
    const float* __restrict__ W2, const float* __restrict__ b2, float* __restrict__ O2,
    const float* __restrict__ W3, const float* __restrict__ b3, float* __restrict__ O3,
    int nw, int n)
{
    __shared__ _Float16 sA[64 * PITCH];     // (normalized) activations, f16
    __shared__ _Float16 sWT[128 * PITCH];   // current weight transposed: sWT[ncol][k]
    const int t = threadIdx.x;
    const int wave = t >> 5;
    const int lane = t & 31;
    const int rowbase = blockIdx.x * 64;

    // ---- stage A tile with fused LayerNorm: one wave32 per row --------------
    for (int r8 = 0; r8 < 8; ++r8) {
        int r  = wave * 8 + r8;
        int gr = rowbase + r;
        if (gr < n) {
            float v[4];
            #pragma unroll
            for (int i = 0; i < 4; ++i) v[i] = A[(size_t)gr * 128 + lane + 32 * i];
            if (doLN) {
                float s = v[0] + v[1] + v[2] + v[3];
                #pragma unroll
                for (int o = 16; o; o >>= 1) s += __shfl_xor(s, o, 32);
                float mu = s * (1.0f / 128.0f);
                float qq = 0.f;
                #pragma unroll
                for (int i = 0; i < 4; ++i) { float d = v[i] - mu; qq += d * d; }
                #pragma unroll
                for (int o = 16; o; o >>= 1) qq += __shfl_xor(qq, o, 32);
                float rs = rsqrtf(qq * (1.0f / 128.0f) + 1e-5f);
                #pragma unroll
                for (int i = 0; i < 4; ++i) {
                    int c = lane + 32 * i;
                    sA[r * PITCH + c] = (_Float16)((v[i] - mu) * rs * lns[c] + lnb[c]);
                }
            } else {
                #pragma unroll
                for (int i = 0; i < 4; ++i)
                    sA[r * PITCH + lane + 32 * i] = (_Float16)v[i];
            }
        } else {
            #pragma unroll
            for (int i = 0; i < 4; ++i)
                sA[r * PITCH + lane + 32 * i] = (_Float16)0.0f;
        }
    }

    const int m  = lane & 15;     // row (A) / col (B,C) within tile
    const int hi = lane >> 4;     // lane half -> K/M sub-block per ISA layout
    const int rstripe = wave & 3;
    const int chalf   = wave >> 2;
    const int arow = rstripe * 16 + m;

    const float* Ws[4] = {W0, W1, W2, W3};
    const float* bs[4] = {b0, b1, b2, b3};
    float*       Os[4] = {O0, O1, O2, O3};

    for (int wsel = 0; wsel < nw; ++wsel) {
        const float* W    = Ws[wsel];
        const float* bias = bs[wsel];
        float*       OUT  = Os[wsel];

        // prefetch next weight into cache while we compute with this one
        if (wsel + 1 < nw)
            __builtin_prefetch(Ws[wsel + 1] + t * 64, 0, 3);  // global_prefetch_b8

        __syncthreads();   // previous compute finished reading sWT (and sA ready)
        for (int i = t; i < 128 * 128; i += 256) {
            int kk = i >> 7, c = i & 127;
            sWT[c * PITCH + kk] = (_Float16)W[kk * 128 + c];
        }
        __syncthreads();

        v8f acc[4];
        #pragma unroll
        for (int nt = 0; nt < 4; ++nt) {
            float b = bias[chalf * 64 + nt * 16 + m];
            #pragma unroll
            for (int j = 0; j < 8; ++j) acc[nt][j] = b;
        }

        for (int kk = 0; kk < 128; kk += 32) {
            // A fragment: 16-bit A 16x32 layout (VGPR j <-> K pair per ISA table)
            union { unsigned u[8]; v16h h; } af;
            #pragma unroll
            for (int j = 0; j < 8; ++j) {
                int koff = ((j >> 2) << 4) + (hi << 3) + ((j & 3) << 1);
                af.u[j] = *(const unsigned*)&sA[arow * PITCH + kk + koff];
            }
            #pragma unroll
            for (int nt = 0; nt < 4; ++nt) {
                int ncol = chalf * 64 + nt * 16 + m;
                // B fragment: 32x16, same K order as A (VGPR j <-> K pair per ISA table)
                union { unsigned u[8]; v16h h; } bf;
                #pragma unroll
                for (int j = 0; j < 8; ++j)
                    bf.u[j] = *(const unsigned*)&sWT[ncol * PITCH + kk + ((j >> 2) << 4) + (hi << 3) + ((j & 3) << 1)];
                acc[nt] = __builtin_amdgcn_wmma_f32_16x16x32_f16(
                    false, af.h, false, bf.h, (short)0, acc[nt], false, false);
            }
        }

        // D layout: VGPR j -> M = j + 8*hi, col = lane&15
        #pragma unroll
        for (int nt = 0; nt < 4; ++nt) {
            int ncol = chalf * 64 + nt * 16 + m;
            #pragma unroll
            for (int j = 0; j < 8; ++j) {
                int gr = rowbase + rstripe * 16 + (hi << 3) + j;
                if (gr < n) OUT[(size_t)gr * 128 + ncol] = acc[nt][j];
            }
        }
    }
}

// ---------------- LayerNorm over D=128, one wave32 per row (final LN) -------
__global__ __launch_bounds__(256) void ln_kernel(
    const float* __restrict__ X, const float* __restrict__ scale,
    const float* __restrict__ bias, float* __restrict__ OUT, int n)
{
    int wid = threadIdx.x >> 5, lane = threadIdx.x & 31;
    int row = blockIdx.x * 8 + wid;
    if (row >= n) return;
    const float* x = X + (size_t)row * 128;
    float v[4], s = 0.f;
    #pragma unroll
    for (int i = 0; i < 4; ++i) { v[i] = x[lane + 32 * i]; s += v[i]; }
    #pragma unroll
    for (int o = 16; o; o >>= 1) s += __shfl_xor(s, o, 32);
    float mu = s * (1.0f / 128.0f);
    float q = 0.f;
    #pragma unroll
    for (int i = 0; i < 4; ++i) { float d = v[i] - mu; q += d * d; }
    #pragma unroll
    for (int o = 16; o; o >>= 1) q += __shfl_xor(q, o, 32);
    float rs = rsqrtf(q * (1.0f / 128.0f) + 1e-5f);
    #pragma unroll
    for (int i = 0; i < 4; ++i) {
        int c = lane + 32 * i;
        OUT[(size_t)row * 128 + c] = (v[i] - mu) * rs * scale[c] + bias[c];
    }
}

// ---------------- degree counting -------------------------------------------
__global__ void deg_kernel(const int* __restrict__ ei, int* din, int* dou, int e)
{
    int i = blockIdx.x * blockDim.x + threadIdx.x;
    if (i >= e) return;
    atomicAdd(&dou[ei[i]], 1);        // src -> out-degree
    atomicAdd(&din[ei[e + i]], 1);    // tgt -> in-degree
}

// ---------------- centrality embedding add ----------------------------------
__global__ void emb_kernel(float* __restrict__ h, const float* __restrict__ in_emb,
                           const float* __restrict__ out_emb,
                           const int* __restrict__ din, const int* __restrict__ dou, int n)
{
    int idx = blockIdx.x * blockDim.x + threadIdx.x;
    if (idx >= n * 128) return;
    int node = idx >> 7, d = idx & 127;
    int a = min(din[node], 512), b = min(dou[node], 512);
    h[idx] += in_emb[a * 128 + d] + out_emb[b * 128 + d];
}

// ---------------- attention buffers init ------------------------------------
__global__ void att_init_kernel(unsigned* __restrict__ mx, float* __restrict__ den,
                                float* __restrict__ agg, int n)
{
    int idx = blockIdx.x * blockDim.x + threadIdx.x;
    if (idx >= n * 128) return;
    agg[idx] = 0.f;
    if (idx < n * 16) { mx[idx] = ENC_NEG_INF; den[idx] = 0.f; }
}

// ---------------- pass1: per-(edge,head) logits + segment max ---------------
__global__ void att_logits_kernel(const float* __restrict__ q, const float* __restrict__ k,
                                  const int* __restrict__ ei, float* __restrict__ alpha,
                                  unsigned* __restrict__ mx, int e)
{
    int idx = blockIdx.x * blockDim.x + threadIdx.x;
    if (idx >= e * 16) return;
    int ed = idx >> 4, hh = idx & 15;
    int s = ei[ed], t = ei[e + ed];
    const float4* qp = (const float4*)(q + (size_t)t * 128 + hh * 8);
    const float4* kp = (const float4*)(k + (size_t)s * 128 + hh * 8);
    float4 q0 = qp[0], q1 = qp[1], k0 = kp[0], k1 = kp[1];
    float a = q0.x * k0.x + q0.y * k0.y + q0.z * k0.z + q0.w * k0.w
            + q1.x * k1.x + q1.y * k1.y + q1.z * k1.z + q1.w * k1.w;
    a *= 0.35355339059327373f;   // 1/sqrt(8)
    alpha[idx] = a;
    atomicMax(&mx[t * 16 + hh], enc_f(a));
}

// ---------------- pass2: exp(alpha - max) + segment sum ---------------------
__global__ void att_exp_kernel(float* __restrict__ alpha, const unsigned* __restrict__ mx,
                               float* __restrict__ den, const int* __restrict__ ei, int e)
{
    int idx = blockIdx.x * blockDim.x + threadIdx.x;
    if (idx >= e * 16) return;
    int ed = idx >> 4, hh = idx & 15;
    int t = ei[e + ed];
    float ex = __expf(alpha[idx] - dec_f(mx[t * 16 + hh]));
    alpha[idx] = ex;
    atomicAdd(&den[t * 16 + hh], ex);
}

// ---------------- pass3: weighted message aggregation -----------------------
__global__ void att_agg_kernel(const float* __restrict__ alpha, const float* __restrict__ den,
                               const float* __restrict__ v, const int* __restrict__ ei,
                               float* __restrict__ agg, int e)
{
    int idx = blockIdx.x * blockDim.x + threadIdx.x;
    if (idx >= e * 16) return;
    int ed = idx >> 4, hh = idx & 15;
    int s = ei[ed], t = ei[e + ed];
    float w = alpha[idx] / den[t * 16 + hh];
    const float* vp = v + (size_t)s * 128 + hh * 8;
    float* ap = agg + (size_t)t * 128 + hh * 8;
    #pragma unroll
    for (int c = 0; c < 8; ++c) atomicAdd(&ap[c], vp[c] * w);
}

// ---------------- beta gate + residual: one wave32 per node ------------------
__global__ __launch_bounds__(256) void beta_kernel(
    float* __restrict__ h, const float* __restrict__ agg, const float* __restrict__ xr,
    const float* __restrict__ Wb, int n)
{
    int wid = threadIdx.x >> 5, lane = threadIdx.x & 31;
    int node = blockIdx.x * 8 + wid;
    if (node >= n) return;
    float ov[4], xv[4], s = 0.f;
    #pragma unroll
    for (int i = 0; i < 4; ++i) {
        int c = lane + 32 * i;
        float o = agg[(size_t)node * 128 + c];
        float x = xr[(size_t)node * 128 + c];
        ov[i] = o; xv[i] = x;
        s += o * Wb[c] + x * Wb[128 + c] + (o - x) * Wb[256 + c];
    }
    #pragma unroll
    for (int o = 16; o; o >>= 1) s += __shfl_xor(s, o, 32);
    float beta = 1.0f / (1.0f + __expf(-s));
    #pragma unroll
    for (int i = 0; i < 4; ++i) {
        int c = lane + 32 * i;
        h[(size_t)node * 128 + c] += beta * xv[i] + (1.0f - beta) * ov[i];
    }
}

// ============================================================================
extern "C" void kernel_launch(void* const* d_in, const int* in_sizes, int n_in,
                              void* d_out, int out_size, void* d_ws, size_t ws_size,
                              hipStream_t stream)
{
    (void)n_in; (void)out_size; (void)ws_size;
    const float* x       = (const float*)d_in[0];
    const int*   ei      = (const int*)  d_in[1];
    const float* W_in    = (const float*)d_in[2];
    const float* b_in    = (const float*)d_in[3];
    const float* in_emb  = (const float*)d_in[4];
    const float* out_emb = (const float*)d_in[5];
    const float* ln_s    = (const float*)d_in[6];
    const float* ln_b    = (const float*)d_in[7];
    const float* Wq      = (const float*)d_in[8];
    const float* bq      = (const float*)d_in[9];
    const float* Wk      = (const float*)d_in[10];
    const float* bk      = (const float*)d_in[11];
    const float* Wv      = (const float*)d_in[12];
    const float* bv      = (const float*)d_in[13];
    const float* Wsk     = (const float*)d_in[14];
    const float* bsk     = (const float*)d_in[15];
    const float* Wbeta   = (const float*)d_in[16];
    const float* fn_s    = (const float*)d_in[17];
    const float* fn_b    = (const float*)d_in[18];

    const int N = in_sizes[0] / 128;
    const int E = in_sizes[1] / 2;
    const int L = in_sizes[6] / 128;   // 3

    // workspace carve-up
    size_t nd = (size_t)N * 128;
    float* h     = (float*)d_ws;
    float* q     = h   + nd;
    float* k     = q   + nd;
    float* v     = k   + nd;
    float* xr    = v   + nd;
    float* agg   = xr  + nd;
    float* alpha = agg + nd;                              // E*16 floats
    unsigned* mx = (unsigned*)(alpha + (size_t)E * 16);   // N*16
    float* den   = (float*)(mx + (size_t)N * 16);         // N*16
    int*   din   = (int*)(den + (size_t)N * 16);          // N
    int*   dou   = din + N;                               // N

    const int TB = 256;
    dim3 blk(TB);
    int gN128  = (int)((nd + TB - 1) / TB);
    int gE16   = (E * 16 + TB - 1) / TB;
    int gE     = (E + TB - 1) / TB;
    int gGemm  = (N + 63) / 64;
    int gRows  = (N + 7) / 8;

    // degrees
    hipMemsetAsync(din, 0, 2 * (size_t)N * sizeof(int), stream);
    deg_kernel<<<gE, blk, 0, stream>>>(ei, din, dou, E);

    // h = x @ W_in + b_in  (WMMA, no LN) ; + centrality embeddings
    fused_ln_gemm4_kernel<<<gGemm, blk, 0, stream>>>(
        x, 0, ln_s, ln_b,
        W_in, b_in, h,  W_in, b_in, h,  W_in, b_in, h,  W_in, b_in, h,
        1, N);
    emb_kernel<<<gN128, blk, 0, stream>>>(h, in_emb, out_emb, din, dou, N);

    for (int l = 0; l < L; ++l) {
        const size_t wo = (size_t)l * 128 * 128;
        const size_t bo = (size_t)l * 128;

        // LN fused with all four projections (q,k,v,skip) — A staged once
        fused_ln_gemm4_kernel<<<gGemm, blk, 0, stream>>>(
            h, 1, ln_s + bo, ln_b + bo,
            Wq  + wo, bq  + bo, q,
            Wk  + wo, bk  + bo, k,
            Wv  + wo, bv  + bo, v,
            Wsk + wo, bsk + bo, xr,
            4, N);

        att_init_kernel  <<<gN128, blk, 0, stream>>>(mx, den, agg, N);
        att_logits_kernel<<<gE16,  blk, 0, stream>>>(q, k, ei, alpha, mx, E);
        att_exp_kernel   <<<gE16,  blk, 0, stream>>>(alpha, mx, den, ei, E);
        att_agg_kernel   <<<gE16,  blk, 0, stream>>>(alpha, den, v, ei, agg, E);

        beta_kernel<<<gRows, blk, 0, stream>>>(h, agg, xr, Wbeta + (size_t)l * 384, N);
    }

    ln_kernel<<<gRows, blk, 0, stream>>>(h, fn_s, fn_b, (float*)d_out, N);
}
